// GATModel_66468913873062
// MI455X (gfx1250) — matrix operand order repaired and checked by reading, not compiled, not run on hardware
//
#include <hip/hip_runtime.h>
#include <hip/hip_bf16.h>
#include <math.h>

#define H_     4
#define D_     64
#define HID_   256
#define IN_    64
#define EXTRA_ 8

typedef __attribute__((ext_vector_type(16))) __bf16 v16bf;
typedef __attribute__((ext_vector_type(8)))  __bf16 v8bf;
typedef __attribute__((ext_vector_type(8)))  float  v8f;

// ---------- helpers ----------
__device__ __forceinline__ unsigned rne_bf16_bits(float x) {
  unsigned u = __float_as_uint(x);
  return (u + 0x7fffu + ((u >> 16) & 1u)) >> 16;   // round-to-nearest-even
}
__device__ __forceinline__ __bf16 f2bf(float x) {
  unsigned short s = (unsigned short)rne_bf16_bits(x);
  return __builtin_bit_cast(__bf16, s);
}
__device__ __forceinline__ unsigned f2key(float f) {     // monotonic uint key
  unsigned u = __float_as_uint(f);
  return (u >> 31) ? ~u : (u | 0x80000000u);
}
__device__ __forceinline__ float key2f(unsigned k) {
  return (k >> 31) ? __uint_as_float(k & 0x7fffffffu) : __uint_as_float(~k);
}
__device__ __forceinline__ float lrelu(float v) { return v > 0.f ? v : 0.2f * v; }

// ---------- elementwise ----------
__global__ void fill_f32(float* p, float v, long long n) {
  long long i = (long long)blockIdx.x * blockDim.x + threadIdx.x;
  if (i < n) p[i] = v;
}
__global__ void fill_u32(unsigned* p, unsigned v, long long n) {
  long long i = (long long)blockIdx.x * blockDim.x + threadIdx.x;
  if (i < n) p[i] = v;
}
// packed fp32 -> bf16 conversion, 2 elements per thread (dword stores)
__global__ void cvt_bf16x2(const float* __restrict__ in, unsigned* __restrict__ out,
                           long long n2) {
  long long i = (long long)blockIdx.x * blockDim.x + threadIdx.x;
  if (i >= n2) return;
  unsigned lo = rne_bf16_bits(in[2 * i]);
  unsigned hi = rne_bf16_bits(in[2 * i + 1]);
  out[i] = (hi << 16) | (lo & 0xffffu);
}
// W[K x Nt] fp32 -> Bt[Nt x K] bf16 (transposed for contiguous-K fragment loads)
__global__ void cvt_w_transpose(const float* __restrict__ W, __bf16* __restrict__ Bt,
                                int K, int Nt) {
  int idx = blockIdx.x * blockDim.x + threadIdx.x;
  if (idx >= K * Nt) return;
  int k = idx / Nt, n = idx % Nt;          // coalesced read over n
  Bt[(size_t)n * K + k] = f2bf(W[idx]);
}
__global__ void bias_relu(float* x, const float* __restrict__ b, long long n) {
  long long i = (long long)blockIdx.x * blockDim.x + threadIdx.x;
  if (i < n) {
    float v = x[i] + b[i % HID_];
    x[i] = v > 0.f ? v : 0.f;
  }
}

// ---------- bf16 WMMA GEMM: C[M x 256] = A[M x K] * B[K x 256] ----------
// Bt is B transposed: [256 x K] bf16, so each lane's 16-element fragment is
// 32 contiguous bytes (two global_load_b128). A fragment: two 16B chunks.
// Software-pipelined one K-step ahead so WMMAs overlap in-flight loads.
// grid.x = M/16; block = 128 (4 waves; wave w -> N tiles 4w..4w+3)
__global__ void __launch_bounds__(128) gemm_bf16_n256(const __bf16* __restrict__ A,
                                                      const __bf16* __restrict__ Bt,
                                                      float* __restrict__ C, int K) {
  const int Nt    = 256;
  const int lane  = threadIdx.x & 31;
  const int wave  = threadIdx.x >> 5;
  const int mtile = blockIdx.x;
  const int row   = mtile * 16 + (lane & 15);
  const int kgrpA = (lane & 16) ? 16 : 0;   // lanes 16-31: K offset +16
  const int kgrpB = (lane & 16) ? 16 : 0;   // lanes 16-31: K offset +16
  const int ncol  = lane & 15;
  const int ntile0 = wave * 4;

  const __bf16* Arow = A + (size_t)row * K + kgrpA;
  const __bf16* B0   = Bt + (size_t)(ntile0 * 16 + ncol) * K + kgrpB;
  const __bf16* B1   = B0 + (size_t)16 * K;
  const __bf16* B2   = B1 + (size_t)16 * K;
  const __bf16* B3   = B2 + (size_t)16 * K;

  auto loadA = [&](int k) -> v16bf {
    v8bf lo = *(const v8bf*)(Arow + k);        // K[k .. k+7]   (pairs 0-3)
    v8bf hi = *(const v8bf*)(Arow + k + 8);    // K[k+8..k+15]  (pairs 4-7)
    return __builtin_shufflevector(lo, hi, 0, 1, 2, 3, 4, 5, 6, 7,
                                   8, 9, 10, 11, 12, 13, 14, 15);
  };

  v8f acc0 = {}, acc1 = {}, acc2 = {}, acc3 = {};

  // prologue: fragments for k0 = 0
  v16bf a  = loadA(0);
  v16bf b0 = *(const v16bf*)(B0);
  v16bf b1 = *(const v16bf*)(B1);
  v16bf b2 = *(const v16bf*)(B2);
  v16bf b3 = *(const v16bf*)(B3);

  int k0 = 0;
  for (; k0 + 32 < K; k0 += 32) {
    const int kn = k0 + 32;
    if (kn + 32 < K) __builtin_prefetch(Arow + kn + 32, 0, 3);   // global_prefetch_b8
    // issue next-iteration loads before consuming current fragments
    v16bf an  = loadA(kn);
    v16bf bn0 = *(const v16bf*)(B0 + kn);
    v16bf bn1 = *(const v16bf*)(B1 + kn);
    v16bf bn2 = *(const v16bf*)(B2 + kn);
    v16bf bn3 = *(const v16bf*)(B3 + kn);

    acc0 = __builtin_amdgcn_wmma_f32_16x16x32_bf16(false, a, false, b0, (short)0, acc0, false, false);
    acc1 = __builtin_amdgcn_wmma_f32_16x16x32_bf16(false, a, false, b1, (short)0, acc1, false, false);
    acc2 = __builtin_amdgcn_wmma_f32_16x16x32_bf16(false, a, false, b2, (short)0, acc2, false, false);
    acc3 = __builtin_amdgcn_wmma_f32_16x16x32_bf16(false, a, false, b3, (short)0, acc3, false, false);

    a = an; b0 = bn0; b1 = bn1; b2 = bn2; b3 = bn3;
  }
  // epilogue: last K-step
  acc0 = __builtin_amdgcn_wmma_f32_16x16x32_bf16(false, a, false, b0, (short)0, acc0, false, false);
  acc1 = __builtin_amdgcn_wmma_f32_16x16x32_bf16(false, a, false, b1, (short)0, acc1, false, false);
  acc2 = __builtin_amdgcn_wmma_f32_16x16x32_bf16(false, a, false, b2, (short)0, acc2, false, false);
  acc3 = __builtin_amdgcn_wmma_f32_16x16x32_bf16(false, a, false, b3, (short)0, acc3, false, false);

  // C/D layout: VGPR r -> rows (mtile*16 + r) lanes 0-15, (mtile*16 + 8 + r) lanes 16-31
  const int mrow = mtile * 16 + ((lane & 16) ? 8 : 0);
  const v8f accs[4] = {acc0, acc1, acc2, acc3};
#pragma unroll
  for (int t = 0; t < 4; ++t) {
    int col = (ntile0 + t) * 16 + ncol;
#pragma unroll
    for (int r = 0; r < 8; ++r)
      C[(size_t)(mrow + r) * Nt + col] = accs[t][r];
  }
}

// ---------- attention projections: el/er [N*H] ----------
__global__ void attn_proj(const float* __restrict__ f, const float* __restrict__ al,
                          const float* __restrict__ ar, float* __restrict__ el,
                          float* __restrict__ er, int N) {
  int idx = blockIdx.x * blockDim.x + threadIdx.x;
  if (idx >= N * H_) return;
  int n = idx >> 2, h = idx & 3;
  const float* fr  = f + (size_t)n * HID_ + h * D_;
  const float* alh = al + h * D_;
  const float* arh = ar + h * D_;
  float sl = 0.f, sr = 0.f;
#pragma unroll 8
  for (int d = 0; d < D_; ++d) { sl += fr[d] * alh[d]; sr += fr[d] * arh[d]; }
  el[idx] = sl; er[idx] = sr;
}

// ---------- edge softmax ----------
__global__ void edge_max(const float* __restrict__ el, const float* __restrict__ er,
                         const int* __restrict__ src, const int* __restrict__ dst,
                         unsigned* __restrict__ mkey, long long EtotH) {
  long long idx = (long long)blockIdx.x * blockDim.x + threadIdx.x;
  if (idx >= EtotH) return;
  long long e = idx >> 2; int h = (int)(idx & 3);
  float v = lrelu(el[(size_t)src[e] * H_ + h] + er[(size_t)dst[e] * H_ + h]);
  atomicMax(&mkey[(size_t)dst[e] * H_ + h], f2key(v));
}
__global__ void edge_exp(const float* __restrict__ el, const float* __restrict__ er,
                         const int* __restrict__ src, const int* __restrict__ dst,
                         const unsigned* __restrict__ mkey, float* __restrict__ ex,
                         float* __restrict__ denom, long long EtotH) {
  long long idx = (long long)blockIdx.x * blockDim.x + threadIdx.x;
  if (idx >= EtotH) return;
  long long e = idx >> 2; int h = (int)(idx & 3);
  int d0 = dst[e];
  float v = lrelu(el[(size_t)src[e] * H_ + h] + er[(size_t)d0 * H_ + h]);
  float xv = expf(v - key2f(mkey[(size_t)d0 * H_ + h]));
  ex[idx] = xv;
  atomicAdd(&denom[(size_t)d0 * H_ + h], xv);
}
// one block (256 threads) per edge: coalesced feature aggregation
__global__ void edge_agg(const float* __restrict__ f, const int* __restrict__ src,
                         const int* __restrict__ dst, const float* __restrict__ ex,
                         const float* __restrict__ denom, float* __restrict__ acc) {
  int e = blockIdx.x, c = threadIdx.x;
  int s = src[e], d0 = dst[e];
  int h = c >> 6;
  float alpha = ex[(size_t)e * H_ + h] / denom[(size_t)d0 * H_ + h];
  atomicAdd(&acc[(size_t)d0 * HID_ + c], alpha * f[(size_t)s * HID_ + c]);
}

// ---------- readout ----------
__global__ void node_score(const float* __restrict__ y, const float* __restrict__ Ws,
                           const float* __restrict__ bs, float* __restrict__ wnode, int N) {
  int n = blockIdx.x * blockDim.x + threadIdx.x;
  if (n >= N) return;
  const float* yr = y + (size_t)n * HID_;
  float s = bs[0];
#pragma unroll 8
  for (int c = 0; c < HID_; ++c) s += yr[c] * Ws[c];
  wnode[n] = 1.f / (1.f + expf(-s));
}
__global__ void scatter_readout(const float* __restrict__ y, const float* __restrict__ wnode,
                                const int* __restrict__ gid, float* __restrict__ num,
                                float* __restrict__ den) {
  int n = blockIdx.x, c = threadIdx.x;
  int g = gid[n];
  float wn = wnode[n];
  atomicAdd(&num[(size_t)g * HID_ + c], wn * y[(size_t)n * HID_ + c]);
  if (c == 0) atomicAdd(&den[g], wn);
}
__global__ void concat_emb(const float* __restrict__ num, const float* __restrict__ den,
                           const float* __restrict__ fg, float* __restrict__ hbuf, int G) {
  int idx = blockIdx.x * blockDim.x + threadIdx.x;
  const int W = HID_ + EXTRA_;
  if (idx >= G * W) return;
  int g = idx / W, c = idx % W;
  if (c < HID_) {
    float d = den[g];
    d = (d == 0.f) ? 1.f : d;
    hbuf[idx] = num[(size_t)g * HID_ + c] / d;
  } else {
    hbuf[idx] = fg[(size_t)g * EXTRA_ + (c - HID_)];
  }
}
// generic dense layer, act: 0=none 1=selu
__global__ void mlp_layer(const float* __restrict__ in, const float* __restrict__ W,
                          const float* __restrict__ b, float* __restrict__ out,
                          int G, int In, int On, int act) {
  int idx = blockIdx.x * blockDim.x + threadIdx.x;
  if (idx >= G * On) return;
  int g = idx / On, j = idx % On;
  const float* ir = in + (size_t)g * In;
  float s = b[j];
  for (int i = 0; i < In; ++i) s += ir[i] * W[(size_t)i * On + j];
  if (act) {
    const float scale = 1.0507009873554805f, alpha = 1.6732632423543772f;
    s = s > 0.f ? scale * s : scale * alpha * (expf(s) - 1.f);
  }
  out[idx] = s;
}

// ---------- launch ----------
extern "C" void kernel_launch(void* const* d_in, const int* in_sizes, int n_in,
                              void* d_out, int out_size, void* d_ws, size_t ws_size,
                              hipStream_t stream) {
  const float* feats_node  = (const float*)d_in[0];
  const float* feats_graph = (const float*)d_in[1];
  const float* W1  = (const float*)d_in[2];
  const float* al1 = (const float*)d_in[3];
  const float* ar1 = (const float*)d_in[4];
  const float* b1  = (const float*)d_in[5];
  const float* W2  = (const float*)d_in[6];
  const float* al2 = (const float*)d_in[7];
  const float* ar2 = (const float*)d_in[8];
  const float* b2  = (const float*)d_in[9];
  const float* Ws  = (const float*)d_in[10];
  const float* bs  = (const float*)d_in[11];
  const float* Wm1 = (const float*)d_in[12];
  const float* bm1 = (const float*)d_in[13];
  const float* Wm2 = (const float*)d_in[14];
  const float* bm2 = (const float*)d_in[15];
  const float* Wm3 = (const float*)d_in[16];
  const float* bm3 = (const float*)d_in[17];
  const int*   src = (const int*)d_in[18];
  const int*   dst = (const int*)d_in[19];
  const int*   gid = (const int*)d_in[20];

  const int N = in_sizes[0] / IN_;         // 100000
  const int G = in_sizes[1] / EXTRA_;      // 2000
  const long long Etot = in_sizes[18];     // E + N

  // workspace carve
  size_t off = 0;
  auto carve = [&](size_t bytes) -> void* {
    void* p = (char*)d_ws + off;
    off += (bytes + 255) & ~(size_t)255;
    return p;
  };
  __bf16*   xb    = (__bf16*)carve((size_t)N * HID_ * 2);
  __bf16*   wb    = (__bf16*)carve((size_t)HID_ * HID_ * 2);   // transposed weights
  float*    f     = (float*)carve((size_t)N * HID_ * 4);
  float*    acc   = (float*)carve((size_t)N * HID_ * 4);       // agg out, then x2/y in place
  float*    el    = (float*)carve((size_t)N * H_ * 4);
  float*    er    = (float*)carve((size_t)N * H_ * 4);
  unsigned* mkey  = (unsigned*)carve((size_t)N * H_ * 4);
  float*    denom = (float*)carve((size_t)N * H_ * 4);
  float*    ex    = (float*)carve((size_t)Etot * H_ * 4);
  float*    wnode = (float*)carve((size_t)N * 4);
  float*    num   = (float*)carve((size_t)G * HID_ * 4);
  float*    den   = (float*)carve((size_t)G * 4);
  float*    hbuf  = (float*)carve((size_t)G * (HID_ + EXTRA_) * 4);
  float*    h1    = (float*)carve((size_t)G * 128 * 4);
  float*    h2    = (float*)carve((size_t)G * 64 * 4);

  const unsigned NEG_INF_KEY = 0x007FFFFFu;  // f2key(-inf)
  auto blks = [](long long n, int b) { return (unsigned)((n + b - 1) / b); };
  const long long NH = (long long)N * H_, NHID = (long long)N * HID_, EH = Etot * H_;

  for (int layer = 0; layer < 2; ++layer) {
    const float* Wl = layer ? W2 : W1;
    const float* al = layer ? al2 : al1;
    const float* ar = layer ? ar2 : ar1;
    const float* bl = layer ? b2 : b1;
    const int    K  = layer ? HID_ : IN_;

    // stage bf16 operands (layer0: A = feats_node; layer1: A = acc, already relu'd)
    const float* Asrc = layer ? acc : feats_node;
    cvt_bf16x2<<<blks((long long)N * K / 2, 256), 256, 0, stream>>>(
        Asrc, (unsigned*)xb, (long long)N * K / 2);
    cvt_w_transpose<<<blks((long long)K * HID_, 256), 256, 0, stream>>>(Wl, wb, K, HID_);

    // f = A @ W  (WMMA bf16 -> f32)
    gemm_bf16_n256<<<dim3(N / 16), 128, 0, stream>>>(xb, wb, f, K);

    // attention projections
    attn_proj<<<blks(NH, 256), 256, 0, stream>>>(f, al, ar, el, er, N);

    // edge softmax + aggregate
    fill_u32<<<blks(NH, 256), 256, 0, stream>>>(mkey, NEG_INF_KEY, NH);
    fill_f32<<<blks(NH, 256), 256, 0, stream>>>(denom, 0.f, NH);
    fill_f32<<<blks(NHID, 256), 256, 0, stream>>>(acc, 0.f, NHID);
    edge_max<<<blks(EH, 256), 256, 0, stream>>>(el, er, src, dst, mkey, EH);
    edge_exp<<<blks(EH, 256), 256, 0, stream>>>(el, er, src, dst, mkey, ex, denom, EH);
    edge_agg<<<(unsigned)Etot, 256, 0, stream>>>(f, src, dst, ex, denom, acc);

    // acc = relu(acc + b)
    bias_relu<<<blks(NHID, 256), 256, 0, stream>>>(acc, bl, NHID);
  }

  // weighted-average readout
  fill_f32<<<blks((long long)G * HID_, 256), 256, 0, stream>>>(num, 0.f, (long long)G * HID_);
  fill_f32<<<blks(G, 256), 256, 0, stream>>>(den, 0.f, G);
  node_score<<<blks(N, 256), 256, 0, stream>>>(acc, Ws, bs, wnode, N);
  scatter_readout<<<(unsigned)N, 256, 0, stream>>>(acc, wnode, gid, num, den);
  concat_emb<<<blks((long long)G * (HID_ + EXTRA_), 256), 256, 0, stream>>>(
      num, den, feats_graph, hbuf, G);

  // MLP 264 -> 128 -> 64 -> 1 (SELU, SELU, none)
  mlp_layer<<<blks((long long)G * 128, 256), 256, 0, stream>>>(hbuf, Wm1, bm1, h1, G, HID_ + EXTRA_, 128, 1);
  mlp_layer<<<blks((long long)G * 64, 256), 256, 0, stream>>>(h1, Wm2, bm2, h2, G, 128, 64, 1);
  mlp_layer<<<blks(G, 256), 256, 0, stream>>>(h2, Wm3, bm3, (float*)d_out, G, 64, 1, 0);
}
